// TaskOrientedPhenotypeLearning_9122510536901
// MI455X (gfx1250) — matrix operand order repaired and checked by reading, not compiled, not run on hardware
//
#include <hip/hip_runtime.h>
#include <hip/hip_bf16.h>

// ---------------- problem constants (match reference setup_inputs) ----------
#define BATCH   2048
#define NVIEW   2
#define DIM     256
#define VB      (BATCH * NVIEW)     // 4096
#define NCAT    10
#define NCONT   10

#define TEMP_F      0.1f
#define INV_TEMP    10.0f
#define BASE_TEMP_F 0.07f
#define ALPHA_F     5.0f

typedef __attribute__((ext_vector_type(16))) _Float16 v16h;
typedef __attribute__((ext_vector_type(8)))  _Float16 v8h;
typedef __attribute__((ext_vector_type(8)))  float    v8f;
typedef __attribute__((ext_vector_type(4)))  unsigned int v4u;
typedef __attribute__((ext_vector_type(8)))  int      v8i_;
typedef __attribute__((ext_vector_type(4)))  int      v4i_;

union AFrag { v16h v; v8h h[2]; };

// ---------------------------------------------------------------------------
// Per-feature phenotype statistics.
// ---------------------------------------------------------------------------
__global__ __launch_bounds__(256)
void stats_kernel(const int* __restrict__ cat, const float* __restrict__ cont,
                  float* __restrict__ cat_conf, float* __restrict__ cont_mean,
                  float* __restrict__ cont_std, float* __restrict__ cont_conf) {
    __shared__ float s0[256], s1[256], s2[256];
    const int f = blockIdx.x;
    const int tid = threadIdx.x;
    float cnt = 0.f, sum = 0.f, sq = 0.f;
    if (f < NCAT) {
        for (int i = tid; i < BATCH; i += 256)
            cnt += (cat[i * NCAT + f] != -1) ? 1.f : 0.f;
    } else {
        const int ff = f - NCAT;
        for (int i = tid; i < BATCH; i += 256) {
            float x = cont[i * NCONT + ff];
            if (x != -1.0f) { cnt += 1.f; sum += x; sq += x * x; }
        }
    }
    s0[tid] = cnt; s1[tid] = sum; s2[tid] = sq;
    __syncthreads();
    for (int s = 128; s > 0; s >>= 1) {
        if (tid < s) { s0[tid] += s0[tid+s]; s1[tid] += s1[tid+s]; s2[tid] += s2[tid+s]; }
        __syncthreads();
    }
    if (tid == 0) {
        const float c = s0[0];
        const float conf = c / (float)BATCH;
        const float gated = (c > 1.f) ? conf : 0.f;   // cnt<=1 -> sim forced to 0
        if (f < NCAT) {
            cat_conf[f] = gated;
        } else {
            const int ff = f - NCAT;
            const float mean = s1[0] / fmaxf(c, 1.f);
            const float var  = (s2[0] - c * mean * mean) / fmaxf(c - 1.f, 1.f);
            cont_mean[ff] = mean;
            cont_std[ff]  = sqrtf(var) + 1e-6f;
            cont_conf[ff] = gated;
        }
    }
}

// ---------------------------------------------------------------------------
// Normalize features -> f16 contrast matrix [VB, DIM]
// contrast[v*B + i, :] = features[i, v, :] / max(||.||, 1e-12)
// ---------------------------------------------------------------------------
__global__ __launch_bounds__(256)
void normalize_kernel(const float* __restrict__ feat, _Float16* __restrict__ cf16) {
    __shared__ float red[256];
    const int a = blockIdx.x;
    const int v = a / BATCH;
    const int i = a % BATCH;
    const int d = threadIdx.x;
    const float x = feat[((size_t)i * NVIEW + v) * DIM + d];
    red[d] = x * x;
    __syncthreads();
    for (int s = 128; s > 0; s >>= 1) {
        if (d < s) red[d] += red[d + s];
        __syncthreads();
    }
    const float inv = 1.0f / fmaxf(sqrtf(red[0]), 1e-12f);
    cf16[(size_t)a * DIM + d] = (_Float16)(x * inv);
}

__global__ __launch_bounds__(256)
void init_kernel(float* __restrict__ p, int n) {
    int idx = blockIdx.x * 256 + threadIdx.x;
    if (idx < n) p[idx] = 0.0f;
}

// ---------------------------------------------------------------------------
// sf[i,j] = alpha_mod(cat_sim) * alpha_mod(cont_sim)
// ---------------------------------------------------------------------------
__device__ __forceinline__ float alpha_mod(float s) {
    return (s > 1.f) ? s * ALPHA_F : ((s < 1.f) ? s / ALPHA_F : s);
}

__global__ __launch_bounds__(256)
void sf_kernel(const int* __restrict__ cat, const float* __restrict__ cont,
               const float* __restrict__ cat_conf, const float* __restrict__ cont_mean,
               const float* __restrict__ cont_std, const float* __restrict__ cont_conf,
               float* __restrict__ sf) {
    const int j = blockIdx.x * 256 + threadIdx.x;
    const int i = blockIdx.y;
    float s = 0.f;
#pragma unroll
    for (int f = 0; f < NCAT; ++f) {
        const int vi = cat[i * NCAT + f], vj = cat[j * NCAT + f];
        if (vi != -1 && vj != -1 && vi == vj) s += cat_conf[f];
    }
    float cs = alpha_mod((1.f + s) / (float)NCAT);

    float s2 = 0.f;
    if (i != j) {
#pragma unroll
        for (int f = 0; f < NCONT; ++f) {
            const float vi = cont[i * NCONT + f], vj = cont[j * NCONT + f];
            if (vi != -1.f && vj != -1.f) {
                const float dz = (vi - vj) / cont_std[f];      // z_i - z_j
                s2 += __expf(-0.5f * dz * dz) * cont_conf[f];
            }
        }
    }
    float ct = alpha_mod((1.f + s2) / (float)NCONT);
    sf[(size_t)i * BATCH + j] = cs * ct;
}

// ---------------------------------------------------------------------------
// TDM: stage a 64-row x 32-half tile of the contrast matrix into LDS.
// D# group0: count=1, lds_addr, global_addr(57b), type=2.
// D# group1: data_size=2B, tensor 256 x 4096 (stride 256), tile 32 x 64.
// Toolchain uses the 6-arg builtin: (v4u, v8i, v4i, v4i, v8i, i32 cpol).
// ---------------------------------------------------------------------------
__device__ __forceinline__ void tdm_load_tile(const _Float16* gtile, unsigned ldsOff) {
    const unsigned long long ga = (unsigned long long)(uintptr_t)gtile;
    v4u g0;
    g0.x = 1u;                                   // count=1, is_restore=0, gather=0
    g0.y = ldsOff;                               // LDS byte address
    g0.z = (unsigned)(ga & 0xFFFFFFFFu);         // global_addr[31:0]
    g0.w = (unsigned)(ga >> 32) | 0x80000000u;   // global_addr[56:32] | type=2
    v8i_ g1;
    g1[0] = 0x00010000;                          // data_size=1 (2 bytes/elem)
    g1[1] = (int)(256u << 16);                   // tensor_dim0 = 256 (bits[79:48])
    g1[2] = (int)(4096u << 16);                  // tensor_dim1 = 4096 (bits[111:80])
    g1[3] = (int)(32u  << 16);                   // tile_dim0 = 32 (bits[127:112])
    g1[4] = 64;                                  // tile_dim1 = 64 (bits[143:128])
    g1[5] = 256;                                 // tensor_dim0_stride = 256
    g1[6] = 0;
    g1[7] = 0;
    const v4i_ z4 = {0, 0, 0, 0};                // groups 2/3 unused (<=2D tensor)
    const v8i_ z8 = {0, 0, 0, 0, 0, 0, 0, 0};
    __builtin_amdgcn_tensor_load_to_lds(g0, g1, z4, z4, z8, 0);
}

// ---------------------------------------------------------------------------
// WMMA GEMM with TDM-staged, double-buffered LDS tiles.
// Workgroup = 4 waves, 64(M) x 64(N) tile; each wave owns 16 rows.
// PASS2=false: per-row max (int-punned atomicMax; values are positive).
// PASS2=true : sum(exp), masked logit sums + positive counts.
// ---------------------------------------------------------------------------
template <bool PASS2>
__global__ __launch_bounds__(128)
void gemm_kernel(const _Float16* __restrict__ cf16,
                 const float* __restrict__ sf,
                 const int* __restrict__ labels,
                 float* __restrict__ rowmax,
                 float* __restrict__ sumexp, float* __restrict__ sum1,
                 float* __restrict__ cnt1,   float* __restrict__ sum2) {
    __shared__ __align__(128) _Float16 ldsA[2][64][32];
    __shared__ __align__(128) _Float16 ldsB[2][64][32];

    const int lane = threadIdx.x & 31;
    const int wv   = threadIdx.x >> 5;        // 0..3
    const int m    = lane & 15;
    const int hi   = lane >> 4;
    const int rowBase = blockIdx.y * 64 + wv * 16;
    const int colBase = blockIdx.x * 64;
    const bool issuer = (threadIdx.x < 32);   // wave 0 drives the TDM

    v8f acc0 = {}, acc1 = {}, acc2 = {}, acc3 = {};

    if (issuer) {
        tdm_load_tile(cf16 + (size_t)blockIdx.y * 64 * DIM, (unsigned)(uintptr_t)&ldsA[0][0][0]);
        tdm_load_tile(cf16 + (size_t)colBase * DIM,         (unsigned)(uintptr_t)&ldsB[0][0][0]);
    }

    const int KSTEPS = DIM / 32;
    for (int k = 0; k < KSTEPS; ++k) {
        const int buf = k & 1;
        if (issuer) {
            if (k + 1 < KSTEPS) {   // prefetch next K-chunk into the other buffer
                const int k1 = (k + 1) * 32;
                tdm_load_tile(cf16 + (size_t)blockIdx.y * 64 * DIM + k1,
                              (unsigned)(uintptr_t)&ldsA[buf ^ 1][0][0]);
                tdm_load_tile(cf16 + (size_t)colBase * DIM + k1,
                              (unsigned)(uintptr_t)&ldsB[buf ^ 1][0][0]);
                __builtin_amdgcn_s_wait_tensorcnt((short)2);  // current buf complete
            } else {
                __builtin_amdgcn_s_wait_tensorcnt((short)0);
            }
        }
        __syncthreads();   // publish LDS tile to all 4 waves

        // A fragment: row (rowBase+m), K contiguous at hi*16
        const _Float16* ar = &ldsA[buf][wv * 16 + m][0];
        AFrag a;
        a.v = *(const v16h*)(ar + hi * 16);
        // B fragments: column (t*16+m) = contrast row, K contiguous at hi*16
        const v16h b0 = *(const v16h*)(&ldsB[buf][ 0 + m][hi * 16]);
        const v16h b1 = *(const v16h*)(&ldsB[buf][16 + m][hi * 16]);
        const v16h b2 = *(const v16h*)(&ldsB[buf][32 + m][hi * 16]);
        const v16h b3 = *(const v16h*)(&ldsB[buf][48 + m][hi * 16]);
        acc0 = __builtin_amdgcn_wmma_f32_16x16x32_f16(false, a.v, false, b0, (short)0, acc0, false, false);
        acc1 = __builtin_amdgcn_wmma_f32_16x16x32_f16(false, a.v, false, b1, (short)0, acc1, false, false);
        acc2 = __builtin_amdgcn_wmma_f32_16x16x32_f16(false, a.v, false, b2, (short)0, acc2, false, false);
        acc3 = __builtin_amdgcn_wmma_f32_16x16x32_f16(false, a.v, false, b3, (short)0, acc3, false, false);

        __syncthreads();   // all waves done reading buf before DMA reuses it
    }

    if (!PASS2) {
#pragma unroll
        for (int r = 0; r < 8; ++r) {
            float v = fmaxf(fmaxf(acc0[r], acc1[r]), fmaxf(acc2[r], acc3[r]));
#pragma unroll
            for (int off = 1; off < 16; off <<= 1)
                v = fmaxf(v, __shfl_xor(v, off, 32));
            if (m == 0) {
                const int row = rowBase + r + hi * 8;
                atomicMax((int*)(rowmax + row), __float_as_int(v * INV_TEMP));
            }
        }
    } else {
#pragma unroll
        for (int r = 0; r < 8; ++r) {
            const int row = rowBase + r + hi * 8;
            const int ri  = row & (BATCH - 1);
            const float rm = rowmax[row];
            const int   li = labels[ri];
            float se = 0.f, s1 = 0.f, c1 = 0.f, s2 = 0.f;
            const float av[4] = { acc0[r], acc1[r], acc2[r], acc3[r] };
#pragma unroll
            for (int t = 0; t < 4; ++t) {
                const int col = colBase + t * 16 + m;
                const int ci  = col & (BATCH - 1);
                const float L  = av[t] * INV_TEMP;
                const float sv = sf[(size_t)ri * BATCH + ci];
                const float lp = (L - rm) * sv;       // scaled, max-shifted logit
                const bool diag = (row == col);
                se += diag ? 0.f : __expf(lp);
                if (!diag && labels[ci] == li) { s1 += lp; c1 += 1.f; }
                if (!diag && ci == ri)         { s2 += lp; }
            }
#pragma unroll
            for (int off = 1; off < 16; off <<= 1) {
                se += __shfl_xor(se, off, 32);
                s1 += __shfl_xor(s1, off, 32);
                c1 += __shfl_xor(c1, off, 32);
                s2 += __shfl_xor(s2, off, 32);
            }
            if (m == 0) {
                atomicAdd(sumexp + row, se);
                atomicAdd(sum1   + row, s1);
                atomicAdd(cnt1   + row, c1);
                atomicAdd(sum2   + row, s2);
            }
        }
    }
}

// ---------------------------------------------------------------------------
// Final reduction over VB rows -> scalar loss (supervised + eye-mask terms)
// ---------------------------------------------------------------------------
__global__ __launch_bounds__(256)
void finalize_kernel(const float* __restrict__ sumexp, const float* __restrict__ sum1,
                     const float* __restrict__ cnt1,   const float* __restrict__ sum2,
                     float* __restrict__ out) {
    __shared__ float r1[256], r2[256];
    const int tid = threadIdx.x;
    float a1 = 0.f, a2 = 0.f;
    for (int row = tid; row < VB; row += 256) {
        const float lse = __logf(sumexp[row]);
        const float p1  = cnt1[row];
        const float p1d = (p1 < 1e-6f) ? 1.f : p1;
        a1 += (sum1[row] - p1 * lse) / p1d;
        const float p2 = (float)(NVIEW - 1);          // eye-mask positives per row
        a2 += (sum2[row] - p2 * lse) / p2;
    }
    r1[tid] = a1; r2[tid] = a2;
    __syncthreads();
    for (int s = 128; s > 0; s >>= 1) {
        if (tid < s) { r1[tid] += r1[tid+s]; r2[tid] += r2[tid+s]; }
        __syncthreads();
    }
    if (tid == 0) {
        const float sc = -(TEMP_F / BASE_TEMP_F) / (float)VB;
        out[0] = sc * r1[0] + sc * r2[0];
    }
}

// ---------------------------------------------------------------------------
extern "C" void kernel_launch(void* const* d_in, const int* in_sizes, int n_in,
                              void* d_out, int out_size, void* d_ws, size_t ws_size,
                              hipStream_t stream) {
    const float* feat   = (const float*)d_in[0];
    const int*   labels = (const int*)d_in[1];
    const int*   cat    = (const int*)d_in[2];
    const float* cont   = (const float*)d_in[3];
    float* out = (float*)d_out;

    char* ws = (char*)d_ws;
    const size_t SF_BYTES = (size_t)BATCH * BATCH * sizeof(float);   // 16 MB
    const size_t CF_BYTES = (size_t)VB * DIM * sizeof(_Float16);     //  2 MB
    float*     sf     = (float*)ws;
    _Float16*  cf16   = (_Float16*)(ws + SF_BYTES);
    float*     rowmax = (float*)(ws + SF_BYTES + CF_BYTES);
    float*     sumexp = rowmax + VB;
    float*     sum1   = sumexp + VB;
    float*     cnt1   = sum1 + VB;
    float*     sum2   = cnt1 + VB;
    float*     cat_conf  = sum2 + VB;
    float*     cont_mean = cat_conf + 16;
    float*     cont_std  = cont_mean + 16;
    float*     cont_conf = cont_std + 16;

    stats_kernel<<<NCAT + NCONT, 256, 0, stream>>>(cat, cont, cat_conf, cont_mean, cont_std, cont_conf);
    normalize_kernel<<<VB, 256, 0, stream>>>(feat, cf16);
    init_kernel<<<(5 * VB + 255) / 256, 256, 0, stream>>>(rowmax, 5 * VB);
    sf_kernel<<<dim3(BATCH / 256, BATCH), 256, 0, stream>>>(cat, cont, cat_conf, cont_mean, cont_std, cont_conf, sf);

    dim3 ggrid(VB / 64, VB / 64);  // 64 x 64 workgroups, 4 waves each
    gemm_kernel<false><<<ggrid, 128, 0, stream>>>(cf16, sf, labels, rowmax, sumexp, sum1, cnt1, sum2);
    gemm_kernel<true ><<<ggrid, 128, 0, stream>>>(cf16, sf, labels, rowmax, sumexp, sum1, cnt1, sum2);

    finalize_kernel<<<1, 256, 0, stream>>>(sumexp, sum1, cnt1, sum2, out);
}
